// SelfAttentionCQKV_51393578664085
// MI455X (gfx1250) — matrix operand order repaired and checked by reading, not compiled, not run on hardware
//
#include <hip/hip_runtime.h>
#include <hip/hip_bf16.h>

typedef __attribute__((ext_vector_type(16))) __bf16 v16bf;
typedef __attribute__((ext_vector_type(8)))  __bf16 v8bf;
typedef __attribute__((ext_vector_type(4)))  __bf16 v4bf;
typedef __attribute__((ext_vector_type(8)))  float  v8f;
typedef __attribute__((ext_vector_type(4)))  unsigned int u32x4;
typedef __attribute__((ext_vector_type(8)))  int i32x8;
typedef __attribute__((ext_vector_type(4)))  int i32x4;

#define SEQ 2048
#define EMB 1024
#define NH  16
#define HD  64

#if __has_builtin(__builtin_amdgcn_tensor_load_to_lds)
#define HAVE_TDM 1
#if __has_include(<hip/amd_detail/amd_gfx1250_TDM.h>)
#define TDM_6ARG 1   // amdgpu-toolchain (therock) 6-arg builtin
#endif
#endif

static __device__ __forceinline__ v16bf cat16(v8bf lo, v8bf hi) {
  return __builtin_shufflevector(lo, hi, 0,1,2,3,4,5,6,7,8,9,10,11,12,13,14,15);
}

#ifdef HAVE_TDM
// 2-D TDM load: tile_d1 rows of tile_d0 bf16 elements, row stride `stride_el`
// elements in global memory, packed [tile_d1][tile_d0] at `lds_addr`.
// D# packing per CDNA5 ISA (tensor dims == tile
// dims -> no OOB path taken; data_size=1 -> 2 bytes).
static __device__ __forceinline__ void tdm_load_2d(unsigned lds_addr,
                                                   const void* gptr,
                                                   int tile_d0, int tile_d1,
                                                   int stride_el) {
  const unsigned long long ga = (unsigned long long)(uintptr_t)gptr;
  u32x4 g0;
  g0[0] = 1u;                                            // count=1, user D#
  g0[1] = lds_addr;                                      // LDS byte address
  g0[2] = (unsigned)(ga & 0xFFFFFFFFu);                  // global_addr[31:0]
  g0[3] = (unsigned)((ga >> 32) & 0x1FFFFFFu) | (2u << 30); // addr[56:32]|type=2
  i32x8 g1;
  g1[0] = (1 << 16);                                     // data_size=1 (2B)
  g1[1] = (tile_d0 & 0xFFFF) << 16;                      // tensor_dim0[15:0]
  g1[2] = ((tile_d0 >> 16) & 0xFFFF) | ((tile_d1 & 0xFFFF) << 16);
  g1[3] = ((tile_d1 >> 16) & 0xFFFF) | ((tile_d0 & 0xFFFF) << 16); // |tile_dim0
  g1[4] = (tile_d1 & 0xFFFF);                            // tile_dim1, tile_dim2=0
  g1[5] = stride_el;                                     // tensor_dim0_stride lo
  g1[6] = 0;
  g1[7] = 0;
  const i32x4 z4 = {0, 0, 0, 0};
#ifdef TDM_6ARG
  const i32x8 z8 = {0, 0, 0, 0, 0, 0, 0, 0};
  __builtin_amdgcn_tensor_load_to_lds(g0, g1, z4, z4, z8, 0);
#else
  __builtin_amdgcn_tensor_load_to_lds(g0, g1, z4, z4, 0);
#endif
}
#endif

// ---------------------------------------------------------------------------
// fp32 -> bf16 conversion (vectorized x4)
// ---------------------------------------------------------------------------
__global__ __launch_bounds__(256) void f2bf_kernel(const float* __restrict__ in,
                                                   __bf16* __restrict__ out, int n4) {
  int i = blockIdx.x * blockDim.x + threadIdx.x;
  if (i < n4) {
    const float4 f = reinterpret_cast<const float4*>(in)[i];
    v4bf o;
    o[0] = (__bf16)f.x; o[1] = (__bf16)f.y; o[2] = (__bf16)f.z; o[3] = (__bf16)f.w;
    reinterpret_cast<v4bf*>(out)[i] = o;
  }
}

// ---------------------------------------------------------------------------
// C[M,N] = A[M,K] * B[N,K]^T  (bf16 in, fp32 accumulate)
// Block: 256 threads = 8 waves (2 m x 4 n); block tile 128x128; wave tile
// 64x32 = 4x2 WMMA accumulators. K tiles (128x32) for A and B are staged in
// LDS by the Tensor Data Mover (wave 0 issues descriptors, double-buffered,
// TENSORcnt + block barrier handoff). Fallback: manual cooperative staging.
// mode 0: write fp32 C row-major; mode 1: scatter bf16 into Q/K/V [b][h][s][d]
// ---------------------------------------------------------------------------
__global__ __launch_bounds__(256) void gemm_bf16_kernel(
    const __bf16* __restrict__ A, const __bf16* __restrict__ B,
    float* __restrict__ Cf,
    __bf16* __restrict__ Qo, __bf16* __restrict__ Ko, __bf16* __restrict__ Vo,
    int M, int N, int K, int mode)
{
  __shared__ __align__(16) __bf16 As[2][128][32];
  __shared__ __align__(16) __bf16 Bs[2][128][32];

  const int lane = threadIdx.x & 31;
  const int wave = threadIdx.x >> 5;
  const int lm = lane & 15;
  const int lh = lane >> 4;
  const int wm = wave & 1;      // 2 wave rows
  const int wn = wave >> 1;     // 4 wave cols
  const int bm = blockIdx.x * 128;
  const int bn = blockIdx.y * 128;

  v8f acc[4][2] = {};

  const int nk = K / 32;

#ifdef HAVE_TDM
  if (wave == 0) {
    tdm_load_2d((unsigned)(uintptr_t)&As[0][0][0], A + (size_t)bm * K, 32, 128, K);
    tdm_load_2d((unsigned)(uintptr_t)&Bs[0][0][0], B + (size_t)bn * K, 32, 128, K);
  }
  for (int t = 0; t < nk; ++t) {
    const int buf = t & 1;
    if (wave == 0) __builtin_amdgcn_s_wait_tensorcnt(0);
    __syncthreads();
    if (wave == 0 && t + 1 < nk) {   // prefetch next K-slab into other buffer
      tdm_load_2d((unsigned)(uintptr_t)&As[buf ^ 1][0][0],
                  A + (size_t)bm * K + (t + 1) * 32, 32, 128, K);
      tdm_load_2d((unsigned)(uintptr_t)&Bs[buf ^ 1][0][0],
                  B + (size_t)bn * K + (t + 1) * 32, 32, 128, K);
    }
    v16bf bfr[2];
#pragma unroll
    for (int j = 0; j < 2; ++j) {
      const __bf16* pb = &Bs[buf][wn * 32 + j * 16 + lm][16 * lh];
      bfr[j] = cat16(*(const v8bf*)pb, *(const v8bf*)(pb + 8));
    }
#pragma unroll
    for (int i = 0; i < 4; ++i) {
      const __bf16* pa = &As[buf][wm * 64 + i * 16 + lm][16 * lh];
      const v16bf af = cat16(*(const v8bf*)pa, *(const v8bf*)(pa + 8));
#pragma unroll
      for (int j = 0; j < 2; ++j)
        acc[i][j] = __builtin_amdgcn_wmma_f32_16x16x32_bf16(
            false, af, false, bfr[j], (short)0, acc[i][j], false, false);
    }
  }
#else
  const int str = threadIdx.x >> 1;          // 0..127: staged row
  const int stc = (threadIdx.x & 1) * 16;    // 0/16: staged K half
  for (int t = 0; t < nk; ++t) {
    __syncthreads();
    {
      const __bf16* ga = A + ((size_t)(bm + str)) * K + t * 32 + stc;
      *(v8bf*)&As[0][str][stc]     = *(const v8bf*)ga;
      *(v8bf*)&As[0][str][stc + 8] = *(const v8bf*)(ga + 8);
      const __bf16* gb = B + ((size_t)(bn + str)) * K + t * 32 + stc;
      *(v8bf*)&Bs[0][str][stc]     = *(const v8bf*)gb;
      *(v8bf*)&Bs[0][str][stc + 8] = *(const v8bf*)(gb + 8);
    }
    __syncthreads();
    v16bf bfr[2];
#pragma unroll
    for (int j = 0; j < 2; ++j) {
      const __bf16* pb = &Bs[0][wn * 32 + j * 16 + lm][16 * lh];
      bfr[j] = cat16(*(const v8bf*)pb, *(const v8bf*)(pb + 8));
    }
#pragma unroll
    for (int i = 0; i < 4; ++i) {
      const __bf16* pa = &As[0][wm * 64 + i * 16 + lm][16 * lh];
      const v16bf af = cat16(*(const v8bf*)pa, *(const v8bf*)(pa + 8));
#pragma unroll
      for (int j = 0; j < 2; ++j)
        acc[i][j] = __builtin_amdgcn_wmma_f32_16x16x32_bf16(
            false, af, false, bfr[j], (short)0, acc[i][j], false, false);
    }
  }
#endif

#pragma unroll
  for (int i = 0; i < 4; ++i) {
#pragma unroll
    for (int j = 0; j < 2; ++j) {
#pragma unroll
      for (int v = 0; v < 8; ++v) {
        const int row = bm + wm * 64 + i * 16 + v + 8 * lh;  // C layout M = v+8*lh
        const int col = bn + wn * 32 + j * 16 + lm;          //          N = lm
        const float val = acc[i][j][v];
        if (mode == 0) {
          Cf[(size_t)row * N + col] = val;
        } else {
          const int part = col >> 10;        // 0=Q 1=K 2=V
          const int f    = col & 1023;
          const int h    = f >> 6;
          const int d    = f & 63;
          const int b    = row >> 11;        // / SEQ
          const int s    = row & 2047;
          __bf16* dst = (part == 0) ? Qo : (part == 1) ? Ko : Vo;
          dst[(((size_t)(b * NH + h)) * SEQ + s) * HD + d] = (__bf16)val;
        }
      }
    }
  }
}

// ---------------------------------------------------------------------------
// Flash attention: grid (SEQ/128, NH, BS); block = 256 threads = 8 waves.
// Each wave owns 16 query rows; K/V tiles (32 kv rows) staged in LDS.
// ---------------------------------------------------------------------------
__global__ __launch_bounds__(256) void attn_kernel(
    const __bf16* __restrict__ Q, const __bf16* __restrict__ K,
    const __bf16* __restrict__ V, __bf16* __restrict__ O)
{
  __shared__ __align__(16) __bf16 Kt[32][64];     // K tile, row-major [kv][d]
  __shared__ __align__(16) __bf16 VT[64][40];     // V tile transposed [d][kv] (+pad)
  __shared__ __align__(16) __bf16 Pw[8][16][32];  // per-wave P scratch

  const int lane = threadIdx.x & 31;
  const int wave = threadIdx.x >> 5;
  const int lm = lane & 15;
  const int lh = lane >> 4;
  const int qb = blockIdx.x;
  const int h  = blockIdx.y;
  const int b  = blockIdx.z;

  const size_t hoff = ((size_t)(b * NH + h)) * SEQ * HD;
  const __bf16* Qh = Q + hoff;
  const __bf16* Kh = K + hoff;
  const __bf16* Vh = V + hoff;

  const int q0w = qb * 128 + wave * 16;

  // Q fragments for the two K-steps over head_dim (d 0..31, 32..63)
  v16bf qf[2];
#pragma unroll
  for (int c = 0; c < 2; ++c) {
    const __bf16* pq = Qh + (size_t)(q0w + lm) * HD + 32 * c + 16 * lh;
    qf[c] = cat16(*(const v8bf*)pq, *(const v8bf*)(pq + 8));
  }

  v8f oacc[4] = {};
  float mi[8], li[8];
#pragma unroll
  for (int v = 0; v < 8; ++v) { mi[v] = -1e30f; li[v] = 0.0f; }

  const int stg_r  = threadIdx.x >> 3;        // 0..31 kv row
  const int stg_d0 = (threadIdx.x & 7) * 8;   // 0..56 d start

  const int ntiles = 4 * qb + 4;              // causal limit for this block
  for (int t = 0; t < ntiles; ++t) {
    const int kv0 = t * 32;
    __syncthreads();
    {
      const v8bf kk = *(const v8bf*)(Kh + (size_t)(kv0 + stg_r) * HD + stg_d0);
      *(v8bf*)&Kt[stg_r][stg_d0] = kk;
      const v8bf vv = *(const v8bf*)(Vh + (size_t)(kv0 + stg_r) * HD + stg_d0);
#pragma unroll
      for (int i = 0; i < 8; ++i) VT[stg_d0 + i][stg_r] = vv[i];
    }
    __syncthreads();
    if (kv0 > q0w + 15) continue;   // wave-uniform skip; barriers already passed

    // S = Q * K^T  (two 16x16 column tiles, K-dim = head_dim = 64)
    v8f s0 = {}, s1 = {};
#pragma unroll
    for (int c = 0; c < 2; ++c) {
      const __bf16* k0p = &Kt[lm][32 * c + 16 * lh];
      const __bf16* k1p = &Kt[16 + lm][32 * c + 16 * lh];
      const v16bf bf0 = cat16(*(const v8bf*)k0p, *(const v8bf*)(k0p + 8));
      const v16bf bf1 = cat16(*(const v8bf*)k1p, *(const v8bf*)(k1p + 8));
      s0 = __builtin_amdgcn_wmma_f32_16x16x32_bf16(false, qf[c], false, bf0, (short)0, s0, false, false);
      s1 = __builtin_amdgcn_wmma_f32_16x16x32_bf16(false, qf[c], false, bf1, (short)0, s1, false, false);
    }

    // Online softmax (fp32). Row r = q0w + 8*lh + v is shared by the 16 lanes
    // of a half-wave -> butterfly reduce with xor 1,2,4,8 (stays in half).
    float p0[8], p1[8];
#pragma unroll
    for (int v = 0; v < 8; ++v) {
      const int row = q0w + 8 * lh + v;
      float x0 = s0[v] * 0.125f;                 // 1/sqrt(64)
      float x1 = s1[v] * 0.125f;
      if (kv0 + lm      > row) x0 = -1e30f;      // causal mask
      if (kv0 + 16 + lm > row) x1 = -1e30f;
      float rm = fmaxf(x0, x1);
#pragma unroll
      for (int off = 1; off < 16; off <<= 1)
        rm = fmaxf(rm, __shfl_xor(rm, off, 32));
      const float mn = fmaxf(mi[v], rm);
      const float al = __expf(mi[v] - mn);
      const float e0 = __expf(x0 - mn);
      const float e1 = __expf(x1 - mn);
      float rs = e0 + e1;
#pragma unroll
      for (int off = 1; off < 16; off <<= 1)
        rs += __shfl_xor(rs, off, 32);
      li[v] = li[v] * al + rs;
      mi[v] = mn;
      p0[v] = e0; p1[v] = e1;
#pragma unroll
      for (int tt = 0; tt < 4; ++tt) oacc[tt][v] *= al;
    }

    // Re-layout P (C-layout -> A-fragment) through per-wave LDS scratch.
#pragma unroll
    for (int v = 0; v < 8; ++v) {
      Pw[wave][8 * lh + v][lm]      = (__bf16)p0[v];
      Pw[wave][8 * lh + v][16 + lm] = (__bf16)p1[v];
    }
    asm volatile("s_wait_dscnt 0" ::: "memory");  // LDS in-order within wave
    const __bf16* pp = &Pw[wave][lm][16 * lh];
    const v16bf pf = cat16(*(const v8bf*)pp, *(const v8bf*)(pp + 8));

    // O += P * V   (4 column tiles over head_dim)
#pragma unroll
    for (int tt = 0; tt < 4; ++tt) {
      const __bf16* pv = &VT[tt * 16 + lm][16 * lh];
      const v16bf vf = cat16(*(const v8bf*)pv, *(const v8bf*)(pv + 8));
      oacc[tt] = __builtin_amdgcn_wmma_f32_16x16x32_bf16(false, pf, false, vf, (short)0, oacc[tt], false, false);
    }
  }

  // Epilogue: O / l, store bf16 into [b][s][E] layout for the output GEMM.
#pragma unroll
  for (int v = 0; v < 8; ++v) {
    const int s = q0w + 8 * lh + v;
    const float inv = 1.0f / li[v];
#pragma unroll
    for (int tt = 0; tt < 4; ++tt) {
      O[((size_t)(b * SEQ + s)) * EMB + h * HD + tt * 16 + lm] =
          (__bf16)(oacc[tt][v] * inv);
    }
  }
}

// ---------------------------------------------------------------------------
extern "C" void kernel_launch(void* const* d_in, const int* in_sizes, int n_in,
                              void* d_out, int out_size, void* d_ws, size_t ws_size,
                              hipStream_t stream) {
  (void)in_sizes; (void)n_in; (void)out_size; (void)ws_size;
  const float* x    = (const float*)d_in[0];
  const float* Win  = (const float*)d_in[1];
  const float* Wout = (const float*)d_in[2];

  const int T = 2 * SEQ;  // 4096 token rows

  char* ws = (char*)d_ws;
  __bf16* xb    = (__bf16*)ws; ws += (size_t)T * EMB * 2;
  __bf16* winb  = (__bf16*)ws; ws += (size_t)3 * EMB * EMB * 2;
  __bf16* woutb = (__bf16*)ws; ws += (size_t)EMB * EMB * 2;
  __bf16* qbuf  = (__bf16*)ws; ws += (size_t)T * EMB * 2;
  __bf16* kbuf  = (__bf16*)ws; ws += (size_t)T * EMB * 2;
  __bf16* vbuf  = (__bf16*)ws; ws += (size_t)T * EMB * 2;
  __bf16* obuf  = (__bf16*)ws; ws += (size_t)T * EMB * 2;

  { int n4 = T * EMB / 4;       f2bf_kernel<<<n4 / 256, 256, 0, stream>>>(x,    xb,    n4); }
  { int n4 = 3 * EMB * EMB / 4; f2bf_kernel<<<n4 / 256, 256, 0, stream>>>(Win,  winb,  n4); }
  { int n4 = EMB * EMB / 4;     f2bf_kernel<<<n4 / 256, 256, 0, stream>>>(Wout, woutb, n4); }

  // QKV projection, scattered into per-head Q/K/V buffers
  gemm_bf16_kernel<<<dim3(T / 128, 3 * EMB / 128), 256, 0, stream>>>(
      xb, winb, nullptr, qbuf, kbuf, vbuf, T, 3 * EMB, EMB, 1);

  // Causal flash attention
  attn_kernel<<<dim3(SEQ / 128, NH, 2), 256, 0, stream>>>(qbuf, kbuf, vbuf, obuf);

  // Output projection -> fp32 d_out
  gemm_bf16_kernel<<<dim3(T / 128, EMB / 128), 256, 0, stream>>>(
      obuf, woutb, (float*)d_out, nullptr, nullptr, nullptr, T, EMB, EMB, 0);
}
